// SNN_257698038117
// MI455X (gfx1250) — compile-verified, matrix operand order checked by reading, not run on hardware
//
#include <hip/hip_runtime.h>

typedef __attribute__((ext_vector_type(16))) _Float16 v16h;
typedef __attribute__((ext_vector_type(8)))  _Float16 v8h;
typedef __attribute__((ext_vector_type(8)))  float    v8f;
typedef __attribute__((ext_vector_type(4)))  float    f4;
typedef __attribute__((ext_vector_type(4)))  int      v4i;
typedef __attribute__((ext_vector_type(4)))  unsigned v4u;
typedef __attribute__((ext_vector_type(2)))  unsigned v2u;

#define THR   0.10315190601276118f
#define BETA1 0.5817305509700217f

// ---------------- dimensions ----------------
#define BATCH 64
#define TSTEPS 23
#define S1 2356            // 19*124
#define S1P 2394           // 19*126  (x-padded by 1 each side)
#define S2 558             // 9*62
#define S2P 576            // 9*64    (x-padded by 1 each side)
#define IC1 17
#define IC2 80
#define K1  969            // 17*19*3
#define K1P 992
#define K1C 31
#define K2  720            // 80*3*3
#define K2P 736
#define K2C 23
#define FCK 7936
#define FCC 248

// ---------------- workspace byte offsets (all 256B aligned) ----------------
#define OFF_SYN1  ((size_t)0)
#define OFF_MEM1  ((size_t)9650176)
#define OFF_SYN2  ((size_t)19300352)
#define OFF_MEM2  ((size_t)28442624)
#define OFF_MEM3  ((size_t)37584896)
#define OFF_MEM4  ((size_t)37617664)
#define OFF_IN1H  ((size_t)37618176)       // [64][17][19][126] f16, borders stay 0
#define OFF_IN2H  ((size_t)42827520)       // [64][80][9][64]  f16, borders stay 0
#define ZERO_BYTES ((size_t)48725760)      // everything above gets zeroed per call
#define OFF_SPK2H ((size_t)48725760)
#define OFF_CUR3  ((size_t)49741568)
#define OFF_CC    ((size_t)49774336)       // shared gate scratch (conv1 & conv2)
#define OFF_W1P   ((size_t)88375040)
#define OFF_W2P   ((size_t)88502016)
#define OFF_FC1P  ((size_t)88878848)

__device__ __forceinline__ float sigm(float z) { return 1.0f / (1.0f + __expf(-z)); }

__device__ __forceinline__ v16h load_frag(const _Float16* p) {
    // halves [k0..k0+7] and [k0+16..k0+23] -> 16-bit A/B WMMA lane layout
    v8h lo = *(const v8h*)(p);
    v8h hi = *(const v8h*)(p + 16);
    return __builtin_shufflevector(lo, hi, 0,1,2,3,4,5,6,7,8,9,10,11,12,13,14,15);
}

// branch-free 16-element im2col gather (conv1): predicated via cndmask, never via exec branches
__device__ __forceinline__ void gather16(const _Float16* __restrict__ Abase, int y,
                                         const int* koff, const unsigned char* kdy,
                                         int kb, v8h& lo, v8h& hi) {
    const v4i* ko = (const v4i*)(koff + kb);
    v4u dv = *(const v4u*)(kdy + kb);
#pragma unroll
    for (int q = 0; q < 4; ++q) {
        v4i o4 = ko[q];
        unsigned dp = dv[q];
#pragma unroll
        for (int e = 0; e < 4; ++e) {
            int dyb = (int)((dp >> (e * 8)) & 255u);          // biased ky; 255 => k >= K1
            bool ok = (unsigned)(y + dyb - 9) < 19u;
            _Float16 v = Abase[ok ? o4[e] : 0];               // clamped addr, always in-bounds
            v = ok ? v : (_Float16)0.f;
            if (q * 4 + e < 8) lo[q * 4 + e] = v; else hi[(q * 4 + e) & 7] = v;
        }
    }
}

// branch-free 8-element im2col gather (conv2)
__device__ __forceinline__ void gather8(const _Float16* __restrict__ Abase, int y,
                                        const int* koff, const unsigned char* kdy,
                                        int kb, v8h& g) {
    const v4i* ko = (const v4i*)(koff + kb);
    v2u dv = *(const v2u*)(kdy + kb);
#pragma unroll
    for (int q = 0; q < 2; ++q) {
        v4i o4 = ko[q];
        unsigned dp = dv[q];
#pragma unroll
        for (int e = 0; e < 4; ++e) {
            int dyb = (int)((dp >> (e * 8)) & 255u);          // biased ky; 255 => k >= K2
            bool ok = (unsigned)(y + dyb - 1) < 9u;
            _Float16 v = Abase[ok ? o4[e] : 0];
            g[q * 4 + e] = ok ? v : (_Float16)0.f;
        }
    }
}

// ---------------- init / packing ----------------
__global__ __launch_bounds__(256) void k_zero(f4* p, int n4) {
    int i = blockIdx.x * 256 + threadIdx.x;
    if (i < n4) { f4 z = {}; p[i] = z; }
}

__global__ __launch_bounds__(256) void k_pack_w1(const float* __restrict__ w, _Float16* __restrict__ wp) {
    int idx = blockIdx.x * 256 + threadIdx.x;          // 31*64*32 = 63488
    int c = idx >> 11, r = idx & 2047;
    int n = r >> 5, kin = r & 31;
    int k = c * 32 + kin;
    float v = 0.f;
    if (k < K1) {
        int ic = k / 57, t = k - ic * 57, ky = t / 3, kx = t - ky * 3;
        v = w[((n * 17 + ic) * 19 + ky) * 3 + kx];
    }
    wp[idx] = (_Float16)v;
}

__global__ __launch_bounds__(256) void k_pack_w2(const float* __restrict__ w, _Float16* __restrict__ wp) {
    int idx = blockIdx.x * 256 + threadIdx.x;          // 23*256*32 = 188416
    int c = idx >> 13, r = idx & 8191;
    int n = r >> 5, kin = r & 31;
    int k = c * 32 + kin;
    float v = 0.f;
    if (k < K2) {
        int ic = k / 9, t = k - ic * 9, ky = t / 3, kx = t - ky * 3;
        v = w[((n * 80 + ic) * 3 + ky) * 3 + kx];
    }
    wp[idx] = (_Float16)v;
}

__global__ __launch_bounds__(256) void k_pack_fc1(const float* __restrict__ w, _Float16* __restrict__ wp) {
    int idx = blockIdx.x * 256 + threadIdx.x;          // 248*128*32 = 1015808
    int c = idx >> 12, r = idx & 4095;
    int n = r >> 5, kin = r & 31;
    wp[idx] = (_Float16)w[n * FCK + c * 32 + kin];
}

__global__ __launch_bounds__(256) void k_pack_x(const float* __restrict__ x, _Float16* __restrict__ in1h, int t) {
    int idx = blockIdx.x * 256 + threadIdx.x;          // 64*2356 = 150784
    int b = idx / S1, s = idx - b * S1;
    int y = s / 124, xc = s - y * 124;
    in1h[(size_t)b * IC1 * S1P + y * 126 + xc + 1] = (_Float16)x[(size_t)(b * TSTEPS + t) * S1 + s];
}

// ---------------- conv1 implicit GEMM: M=150784 N=64 K=969 ----------------
__global__ __launch_bounds__(128) void k_conv1(const _Float16* __restrict__ A,
                                               const _Float16* __restrict__ Bp,
                                               float* __restrict__ C) {
    __shared__ _Float16 At[2][64 * 32];
    __shared__ int koff[K1P];
    __shared__ unsigned char kdy[K1P];
    int tid = threadIdx.x;
    for (int k = tid; k < K1P; k += 128) {
        int off = 0, dyb = 255;
        if (k < K1) {
            int ic = k / 57, t = k - ic * 57, ky = t / 3, kx = t - ky * 3;
            dyb = ky;                                   // dy = ky-9
            off = ic * S1P + (ky - 9) * 126 + (kx - 1);
        }
        koff[k] = off; kdy[k] = (unsigned char)dyb;
    }
    int r = tid >> 1, h = tid & 1;
    int m = blockIdx.x * 64 + r;
    int b = m / S1, s = m - b * S1;
    int y = s / 124, xc = s - y * 124;
    const _Float16* Abase = A + (size_t)b * IC1 * S1P + y * 126 + (xc + 1);

    int wv = tid >> 5, lane = tid & 31;
    int lrow = lane & 15, hi8 = (lane >> 4) * 8;
    v8f a0c = {}, a1c = {}, a2c = {}, a3c = {};
    __syncthreads();                                     // tables ready

    v8h glo, ghi;
    gather16(Abase, y, koff, kdy, h * 16, glo, ghi);     // chunk 0

    for (int c = 0; c < K1C; ++c) {
        v8h* dst = (v8h*)&At[c & 1][r * 32 + h * 16];
        dst[0] = glo; dst[1] = ghi;
        __syncthreads();
        if (c + 1 < K1C)                                  // prefetch gather of next chunk into regs
            gather16(Abase, y, koff, kdy, (c + 1) * 32 + h * 16, glo, ghi);
        const _Float16* bp = Bp + (size_t)(c * 64 + wv * 16 + lrow) * 32 + hi8;
        if (c + 1 < K1C) __builtin_prefetch(bp + 64 * 32, 0, 0);
        v16h bf = load_frag(bp);
        const _Float16* atb = &At[c & 1][0];
        v16h fa0 = load_frag(&atb[( 0 + lrow) * 32 + hi8]);
        v16h fa1 = load_frag(&atb[(16 + lrow) * 32 + hi8]);
        v16h fa2 = load_frag(&atb[(32 + lrow) * 32 + hi8]);
        v16h fa3 = load_frag(&atb[(48 + lrow) * 32 + hi8]);
        a0c = __builtin_amdgcn_wmma_f32_16x16x32_f16(false, fa0, false, bf, (short)0, a0c, false, false);
        a1c = __builtin_amdgcn_wmma_f32_16x16x32_f16(false, fa1, false, bf, (short)0, a1c, false, false);
        a2c = __builtin_amdgcn_wmma_f32_16x16x32_f16(false, fa2, false, bf, (short)0, a2c, false, false);
        a3c = __builtin_amdgcn_wmma_f32_16x16x32_f16(false, fa3, false, bf, (short)0, a3c, false, false);
    }
    int n = wv * 16 + lrow;
    int mbase = blockIdx.x * 64 + hi8;
#pragma unroll
    for (int v = 0; v < 8; ++v) {
        C[(size_t)(mbase +  0 + v) * 64 + n] = a0c[v];
        C[(size_t)(mbase + 16 + v) * 64 + n] = a1c[v];
        C[(size_t)(mbase + 32 + v) * 64 + n] = a2c[v];
        C[(size_t)(mbase + 48 + v) * 64 + n] = a3c[v];
    }
}

// ---------------- conv2 implicit GEMM: M=35712 N=256 K=720 ----------------
__global__ __launch_bounds__(256) void k_conv2(const _Float16* __restrict__ A,
                                               const _Float16* __restrict__ Bp,
                                               float* __restrict__ C) {
    __shared__ _Float16 At[2][64 * 32];
    __shared__ int koff[K2P];
    __shared__ unsigned char kdy[K2P];
    int tid = threadIdx.x;
    for (int k = tid; k < K2P; k += 256) {
        int off = 0, dyb = 255;
        if (k < K2) {
            int ic = k / 9, t = k - ic * 9, ky = t / 3, kx = t - ky * 3;
            dyb = ky;                                    // dy = ky-1
            off = ic * S2P + (ky - 1) * 64 + (kx - 1);
        }
        koff[k] = off; kdy[k] = (unsigned char)dyb;
    }
    int r = tid >> 2, q = tid & 3;
    int m = blockIdx.x * 64 + r;
    int b = m / S2, s = m - b * S2;
    int y = s / 62, xc = s - y * 62;
    const _Float16* Abase = A + (size_t)b * IC2 * S2P + y * 64 + (xc + 1);

    int wv = tid >> 5, lane = tid & 31;
    int lrow = lane & 15, hi8 = (lane >> 4) * 8;
    v8f acc0[4] = {{}, {}, {}, {}};
    v8f acc1[4] = {{}, {}, {}, {}};
    __syncthreads();

    v8h g;
    gather8(Abase, y, koff, kdy, q * 8, g);              // chunk 0

    for (int c = 0; c < K2C; ++c) {
        *(v8h*)&At[c & 1][r * 32 + q * 8] = g;
        __syncthreads();
        if (c + 1 < K2C)
            gather8(Abase, y, koff, kdy, (c + 1) * 32 + q * 8, g);
        const _Float16* bp0 = Bp + (size_t)(c * 256 + wv * 32 + lrow) * 32 + hi8;
        const _Float16* bp1 = bp0 + 16 * 32;
        if (c + 1 < K2C) __builtin_prefetch(bp0 + 256 * 32, 0, 0);
        v16h bf0 = load_frag(bp0);
        v16h bf1 = load_frag(bp1);
        const _Float16* atb = &At[c & 1][0];
#pragma unroll
        for (int mi = 0; mi < 4; ++mi) {
            v16h fa = load_frag(&atb[(mi * 16 + lrow) * 32 + hi8]);
            acc0[mi] = __builtin_amdgcn_wmma_f32_16x16x32_f16(false, fa, false, bf0, (short)0, acc0[mi], false, false);
            acc1[mi] = __builtin_amdgcn_wmma_f32_16x16x32_f16(false, fa, false, bf1, (short)0, acc1[mi], false, false);
        }
    }
    int n0 = wv * 32 + lrow;
    int mbase = blockIdx.x * 64 + hi8;
#pragma unroll
    for (int mi = 0; mi < 4; ++mi)
#pragma unroll
        for (int v = 0; v < 8; ++v) {
            C[(size_t)(mbase + mi * 16 + v) * 256 + n0]      = acc0[mi][v];
            C[(size_t)(mbase + mi * 16 + v) * 256 + n0 + 16] = acc1[mi][v];
        }
}

// ---------------- fc1 GEMM: M=64 N=128 K=7936, one workgroup ----------------
__global__ __launch_bounds__(256) void k_fc1(const _Float16* __restrict__ A,
                                             const _Float16* __restrict__ Bp,
                                             float* __restrict__ C) {
    int tid = threadIdx.x;
    int wv = tid >> 5, lane = tid & 31;
    int lrow = lane & 15, hi8 = (lane >> 4) * 8;
    int n0 = wv * 16;
    v8f acc[4] = {{}, {}, {}, {}};
    for (int c = 0; c < FCC; ++c) {
        const _Float16* bp = Bp + (size_t)(c * 128 + n0 + lrow) * 32 + hi8;
        if (c + 1 < FCC) __builtin_prefetch(bp + 128 * 32, 0, 0);
        v16h bf = load_frag(bp);
#pragma unroll
        for (int mi = 0; mi < 4; ++mi) {
            const _Float16* ap = A + (size_t)(mi * 16 + lrow) * FCK + c * 32 + hi8;
            v16h fa = load_frag(ap);
            acc[mi] = __builtin_amdgcn_wmma_f32_16x16x32_f16(false, fa, false, bf, (short)0, acc[mi], false, false);
        }
    }
#pragma unroll
    for (int mi = 0; mi < 4; ++mi)
#pragma unroll
        for (int v = 0; v < 8; ++v)
            C[(size_t)(mi * 16 + v + hi8) * 128 + n0 + lrow] = acc[mi][v];
}

// ---------------- elementwise LSTM / pool / head ----------------
__global__ __launch_bounds__(256) void k_update1(const float* __restrict__ cc, const float* __restrict__ bias,
                                                 float* __restrict__ syn, float* __restrict__ mem,
                                                 _Float16* __restrict__ in1h) {
    int idx = blockIdx.x * 256 + threadIdx.x;          // 64*16*2356
    int b = idx / (16 * S1), r = idx - b * (16 * S1);
    int hc = r / S1, s = r - hc * S1;
    int y = s / 124, xc = s - y * 124;
    const float* g = cc + (size_t)(b * S1 + s) * 64;
    float gi = g[hc]      + bias[hc];
    float gf = g[16 + hc] + bias[16 + hc];
    float gg = g[32 + hc] + bias[32 + hc];
    float go = g[48 + hc] + bias[48 + hc];
    float sy = sigm(gf) * syn[idx] + sigm(gi) * tanhf(gg);
    float mm = sigm(go) * tanhf(sy);
    syn[idx] = sy; mem[idx] = mm;
    in1h[(size_t)(b * IC1 + 1 + hc) * S1P + y * 126 + xc + 1] = (_Float16)mm;
}

__global__ __launch_bounds__(256) void k_pool1(const float* __restrict__ mem, _Float16* __restrict__ in2h) {
    int idx = blockIdx.x * 256 + threadIdx.x;          // 64*16*558
    int b = idx / (16 * S2), r = idx - b * (16 * S2);
    int hc = r / S2, s = r - hc * S2;
    int y = s / 62, xc = s - y * 62;
    const float* p = mem + (size_t)(b * 16 + hc) * S1 + 2 * y * 124 + 2 * xc;
    float mx = fmaxf(fmaxf(p[0], p[1]), fmaxf(p[124], p[125]));
    in2h[(size_t)(b * IC2 + hc) * S2P + y * 64 + xc + 1] = (_Float16)((mx - THR) > 0.f ? 1.f : 0.f);
}

__global__ __launch_bounds__(256) void k_update2(const float* __restrict__ cc, const float* __restrict__ bias,
                                                 float* __restrict__ syn, float* __restrict__ mem,
                                                 _Float16* __restrict__ in2h) {
    int idx = blockIdx.x * 256 + threadIdx.x;          // 64*64*558
    int b = idx / (64 * S2), r = idx - b * (64 * S2);
    int c = r / S2, s = r - c * S2;
    int y = s / 62, xc = s - y * 62;
    const float* g = cc + (size_t)(b * S2 + s) * 256;
    float gi = g[c]       + bias[c];
    float gf = g[64 + c]  + bias[64 + c];
    float gg = g[128 + c] + bias[128 + c];
    float go = g[192 + c] + bias[192 + c];
    float sy = sigm(gf) * syn[idx] + sigm(gi) * tanhf(gg);
    float mm = sigm(go) * tanhf(sy);
    syn[idx] = sy; mem[idx] = mm;
    in2h[(size_t)(b * IC2 + 16 + c) * S2P + y * 64 + xc + 1] = (_Float16)mm;
}

__global__ __launch_bounds__(256) void k_pool2(const float* __restrict__ mem, _Float16* __restrict__ spk2) {
    int idx = blockIdx.x * 256 + threadIdx.x;          // 64*64*124 = 507904
    int b = idx / FCK, r = idx - b * FCK;              // r = c*124 + y*31 + x
    int c = r / 124, s = r - c * 124;
    int y = s / 31, xc = s - y * 31;
    const float* p = mem + (size_t)(b * 64 + c) * S2 + 2 * y * 62 + 2 * xc;
    float mx = fmaxf(fmaxf(p[0], p[1]), fmaxf(p[62], p[63]));
    spk2[idx] = (_Float16)((mx - THR) > 0.f ? 1.f : 0.f);
}

__global__ __launch_bounds__(256) void k_head(const float* __restrict__ cur3, const float* __restrict__ fb1,
                                              const float* __restrict__ fw2, const float* __restrict__ fb2,
                                              float* __restrict__ mem3, float* __restrict__ mem4,
                                              float* __restrict__ out, int t) {
    __shared__ float spk3[8192];
    int tid = threadIdx.x;
    for (int i = tid; i < 8192; i += 256) {
        int n = i & 127;
        float cur = cur3[i] + fb1[n];
        float m = mem3[i];
        float rst = (m > THR) ? THR : 0.f;
        m = BETA1 * m + cur - rst;
        mem3[i] = m;
        spk3[i] = ((m - THR) > 0.f) ? 1.f : 0.f;
    }
    __syncthreads();
    if (tid < 128) {
        int b = tid >> 1, j = tid & 1;
        float sum = fb2[j];
        for (int k = 0; k < 128; ++k) sum += spk3[b * 128 + k] * fw2[j * 128 + k];
        float m = mem4[tid];
        float rst = (m > THR) ? THR : 0.f;
        m = 0.5f * m + sum - rst;
        mem4[tid] = m;
        out[t * 128 + tid] = ((m - THR) > 0.f) ? 1.f : 0.f;
        out[TSTEPS * 128 + t * 128 + tid] = m;
    }
}

extern "C" void kernel_launch(void* const* d_in, const int* in_sizes, int n_in,
                              void* d_out, int out_size, void* d_ws, size_t ws_size,
                              hipStream_t stream) {
    (void)in_sizes; (void)n_in; (void)out_size; (void)ws_size;
    const float* x   = (const float*)d_in[0];
    const float* w1  = (const float*)d_in[1];
    const float* b1  = (const float*)d_in[2];
    const float* w2  = (const float*)d_in[3];
    const float* b2  = (const float*)d_in[4];
    const float* fw1 = (const float*)d_in[5];
    const float* fb1 = (const float*)d_in[6];
    const float* fw2 = (const float*)d_in[7];
    const float* fb2 = (const float*)d_in[8];
    float* out = (float*)d_out;
    char* ws = (char*)d_ws;

    float*    syn1  = (float*)(ws + OFF_SYN1);
    float*    mem1  = (float*)(ws + OFF_MEM1);
    float*    syn2  = (float*)(ws + OFF_SYN2);
    float*    mem2  = (float*)(ws + OFF_MEM2);
    float*    mem3  = (float*)(ws + OFF_MEM3);
    float*    mem4  = (float*)(ws + OFF_MEM4);
    _Float16* in1h  = (_Float16*)(ws + OFF_IN1H);
    _Float16* in2h  = (_Float16*)(ws + OFF_IN2H);
    _Float16* spk2h = (_Float16*)(ws + OFF_SPK2H);
    float*    cur3  = (float*)(ws + OFF_CUR3);
    float*    cc    = (float*)(ws + OFF_CC);
    _Float16* w1p   = (_Float16*)(ws + OFF_W1P);
    _Float16* w2p   = (_Float16*)(ws + OFF_W2P);
    _Float16* fc1p  = (_Float16*)(ws + OFF_FC1P);

    int n4 = (int)(ZERO_BYTES / 16);
    k_zero<<<(n4 + 255) / 256, 256, 0, stream>>>((f4*)ws, n4);
    k_pack_w1<<<248, 256, 0, stream>>>(w1, w1p);
    k_pack_w2<<<736, 256, 0, stream>>>(w2, w2p);
    k_pack_fc1<<<3968, 256, 0, stream>>>(fw1, fc1p);

    for (int t = 0; t < TSTEPS; ++t) {
        k_pack_x<<<589, 256, 0, stream>>>(x, in1h, t);
        k_conv1<<<2356, 128, 0, stream>>>(in1h, w1p, cc);
        k_update1<<<9424, 256, 0, stream>>>(cc, b1, syn1, mem1, in1h);
        k_pool1<<<2232, 256, 0, stream>>>(mem1, in2h);
        k_conv2<<<558, 256, 0, stream>>>(in2h, w2p, cc);
        k_update2<<<8928, 256, 0, stream>>>(cc, b2, syn2, mem2, in2h);
        k_pool2<<<1984, 256, 0, stream>>>(mem2, spk2h);
        k_fc1<<<1, 256, 0, stream>>>(spk2h, fc1p, cur3);
        k_head<<<1, 256, 0, stream>>>(cur3, fb1, fw2, fb2, mem3, mem4, out, t);
    }
}
